// DetectionTransformerDecoder_48567490183302
// MI455X (gfx1250) — matrix operand order repaired and checked by reading, not compiled, not run on hardware
//
#include <hip/hip_runtime.h>
#include <hip/hip_bf16.h>
#include <cstdint>

// ---- problem constants ----
#define NLAYER 6
#define C_DIM  256
#define HEADS  8
#define NQ     900
#define BATCH  4
#define HFM    200
#define WFM    200
#define HW     (HFM*WFM)
#define NPTS   8
#define DH     32
#define ROWS   (BATCH*NQ)          // 3600
#define VROWS  (BATCH*HW)          // 160000

typedef __attribute__((ext_vector_type(16))) __bf16 v16bf;
typedef __attribute__((ext_vector_type(8)))  float  v8f;

// f32 -> bf16, round-half-up (32-bit ops only)
__device__ __forceinline__ unsigned short f2bf(float f) {
  return (unsigned short)((__float_as_uint(f) + 0x8000u) >> 16);
}
// pack two floats into one dword of bf16 pairs (lo=a, hi=b) via v_perm_b32
__device__ __forceinline__ unsigned int pack2bf(float a, float b) {
  unsigned int ua = __float_as_uint(a) + 0x8000u;
  unsigned int ub = __float_as_uint(b) + 0x8000u;
  // dst[15:0] = ua[31:16], dst[31:16] = ub[31:16]
  return __builtin_amdgcn_perm(ub, ua, 0x07060302u);
}
__device__ __forceinline__ float bf2f(unsigned short s) {
  return __uint_as_float(((unsigned int)s) << 16);
}
__device__ __forceinline__ float wred32(float x) {   // wave32 sum
  x += __shfl_xor(x, 16); x += __shfl_xor(x, 8); x += __shfl_xor(x, 4);
  x += __shfl_xor(x, 2);  x += __shfl_xor(x, 1);
  return x;
}

union FragU { v16bf v; unsigned int u[8]; };
union AccU  { v8f   v; float        f[8]; };

// =====================================================================
// bf16 WMMA GEMM:  out[M,N] = A[M,K] @ W[N,K]^T + bias (+res) (relu)
// block = 256 threads (8 waves). Tile 256(M) x 64(N); K step 32.
// Register-staged pipeline + double-buffered LDS; 1 barrier per K-step.
// Each wave: 2 A-frags x 4 B-frags = 8 WMMAs per K-step.
// =====================================================================
template<int OUT_BF16, int RELU, int HAS_RES>
__global__ __launch_bounds__(256)
void gemm_bf16_kernel(const float* __restrict__ A,
                      const unsigned short* __restrict__ W,
                      const float* __restrict__ bias,
                      const float* __restrict__ res,
                      void* __restrict__ out,
                      int M, int N, int K) {
  __shared__ unsigned short sA[2][256 * 32];   // [row][k]
  __shared__ unsigned short sB[2][64 * 32];    // [n][k]
  const int t    = threadIdx.x;
  const int wave = t >> 5;
  const int lane = t & 31;
  const int tileM = blockIdx.y * 256;
  const int tileN = blockIdx.x * 64;
  const bool full = (tileM + 256 <= M);

  // staging geometry
  const int aq  = t & 7;    // A col quad (4 floats)
  const int arb = t >> 3;   // A row base 0..31 (rows arb + 32*i)
  const int brow = t >> 2;  // B row 0..63
  const int bq   = t & 3;   // B uint4 quad

  float4 ra[8];
  uint4  rb;

  auto load_tile = [&](int kb) {
    const float* Ap = A + (size_t)(tileM + arb) * K + kb + aq * 4;
    if (full) {
#pragma unroll
      for (int i = 0; i < 8; ++i)
        ra[i] = *(const float4*)(Ap + (size_t)32 * i * K);
    } else {
#pragma unroll
      for (int i = 0; i < 8; ++i) {
        if (tileM + arb + 32 * i < M)
          ra[i] = *(const float4*)(Ap + (size_t)32 * i * K);
        else
          ra[i] = make_float4(0.f, 0.f, 0.f, 0.f);
      }
    }
    rb = *(const uint4*)((const unsigned int*)(W + (size_t)(tileN + brow) * K +
                                               kb) + bq * 4);
  };
  auto store_tile = [&](int buf) {
#pragma unroll
    for (int i = 0; i < 8; ++i) {
      uint2 p = make_uint2(pack2bf(ra[i].x, ra[i].y),
                           pack2bf(ra[i].z, ra[i].w));
      *(uint2*)((unsigned int*)sA[buf] + (arb + 32 * i) * 16 + aq * 2) = p;
    }
    *(uint4*)((unsigned int*)sB[buf] + brow * 16 + bq * 4) = rb;
  };

  AccU acc[2][4];
#pragma unroll
  for (int s = 0; s < 2; ++s)
#pragma unroll
    for (int nt = 0; nt < 4; ++nt)
#pragma unroll
      for (int r = 0; r < 8; ++r) acc[s][nt].f[r] = 0.f;

  const int mrow  = lane & 15;
  const int khalf = lane >> 4;

  load_tile(0);
  int buf = 0;
  for (int kb = 0; kb < K; kb += 32) {
    store_tile(buf);
    __syncthreads();
    if (kb + 32 < K) load_tile(kb + 32);   // next slab in flight during WMMAs

    // A fragments (ISA 16-bit 16x32 layout): rows wave*32 + s*16 + mrow
    FragU af[2];
#pragma unroll
    for (int s = 0; s < 2; ++s) {
      const unsigned int* pa =
          (const unsigned int*)(sA[buf] + (wave * 32 + s * 16 + mrow) * 32);
#pragma unroll
      for (int p = 0; p < 8; ++p) {
        int k0 = 2 * p + (p >= 4 ? 8 : 0) + 8 * khalf;
        af[s].u[p] = pa[k0 >> 1];
      }
    }
    // B fragments + 8 WMMAs
#pragma unroll
    for (int nt = 0; nt < 4; ++nt) {
      FragU bf;
      const unsigned int* pb =
          (const unsigned int*)(sB[buf] + (nt * 16 + mrow) * 32);
#pragma unroll
      for (int p = 0; p < 8; ++p) {
        int k0 = 2 * p + (p >= 4 ? 8 : 0) + 8 * khalf;
        bf.u[p] = pb[k0 >> 1];
      }
      acc[0][nt].v = __builtin_amdgcn_wmma_f32_16x16x32_bf16(
          false, af[0].v, false, bf.v, (short)0, acc[0][nt].v, false, false);
      acc[1][nt].v = __builtin_amdgcn_wmma_f32_16x16x32_bf16(
          false, af[1].v, false, bf.v, (short)0, acc[1][nt].v, false, false);
    }
    buf ^= 1;
  }

  // ---- epilogue ----
  if (full) {
#pragma unroll
    for (int s = 0; s < 2; ++s)
#pragma unroll
      for (int nt = 0; nt < 4; ++nt) {
        int col = tileN + nt * 16 + mrow;
#pragma unroll
        for (int r = 0; r < 8; ++r) {
          int row = tileM + wave * 32 + s * 16 + r + 8 * khalf;
          float v = acc[s][nt].f[r] + bias[col];
          if (HAS_RES) v += res[(size_t)row * N + col];
          if (RELU) v = fmaxf(v, 0.f);
          if (OUT_BF16)
            ((unsigned short*)out)[(size_t)row * N + col] = f2bf(v);
          else
            ((float*)out)[(size_t)row * N + col] = v;
        }
      }
  } else {
#pragma unroll
    for (int s = 0; s < 2; ++s)
#pragma unroll
      for (int nt = 0; nt < 4; ++nt) {
        int col = tileN + nt * 16 + mrow;
#pragma unroll
        for (int r = 0; r < 8; ++r) {
          int row = tileM + wave * 32 + s * 16 + r + 8 * khalf;
          if (row < M) {
            float v = acc[s][nt].f[r] + bias[col];
            if (HAS_RES) v += res[(size_t)row * N + col];
            if (RELU) v = fmaxf(v, 0.f);
            if (OUT_BF16)
              ((unsigned short*)out)[(size_t)row * N + col] = f2bf(v);
            else
              ((float*)out)[(size_t)row * N + col] = v;
          }
        }
      }
  }
}

// =====================================================================
// f32 -> bf16 bulk convert (weights)
// =====================================================================
__global__ void cvt_bf16_kernel(const float* __restrict__ in,
                                unsigned short* __restrict__ out, size_t n) {
  size_t i = (size_t)blockIdx.x * blockDim.x + threadIdx.x;
  if (i < n) out[i] = f2bf(in[i]);
}

// =====================================================================
// value (B,C,HW) -> value_flat (B,HW,C), tiled transpose
// =====================================================================
__global__ void transpose_value_kernel(const float* __restrict__ in,
                                       float* __restrict__ out) {
  __shared__ float tile[32][33];
  int b  = blockIdx.z;
  int c0 = blockIdx.y * 32;
  int s0 = blockIdx.x * 32;
  int tx = threadIdx.x, ty = threadIdx.y;          // (32, 8)
#pragma unroll
  for (int k = 0; k < 4; ++k)
    tile[ty + 8 * k][tx] =
        in[((size_t)b * C_DIM + c0 + ty + 8 * k) * HW + s0 + tx];
  __syncthreads();
#pragma unroll
  for (int k = 0; k < 4; ++k)
    out[((size_t)b * HW + s0 + ty + 8 * k) * C_DIM + c0 + tx] =
        tile[tx][ty + 8 * k];
}

// =====================================================================
// elementwise add: out = a + b
// =====================================================================
__global__ void add_kernel(const float* __restrict__ a,
                           const float* __restrict__ b,
                           float* __restrict__ out, int n) {
  int i = blockIdx.x * blockDim.x + threadIdx.x;
  if (i < n) out[i] = a[i] + b[i];
}

// =====================================================================
// LayerNorm over C=256; optional second input (x+x2); optional 2nd output
// =====================================================================
__global__ __launch_bounds__(256)
void ln_kernel(const float* __restrict__ x, const float* __restrict__ x2,
               const float* __restrict__ g, const float* __restrict__ b,
               float* __restrict__ out, float* __restrict__ out2) {
  __shared__ float sh[16];
  int row = blockIdx.x;
  int t = threadIdx.x, wave = t >> 5, lane = t & 31;
  float v = x[(size_t)row * C_DIM + t];
  if (x2) v += x2[(size_t)row * C_DIM + t];
  float s = wred32(v), q = wred32(v * v);
  if (lane == 0) { sh[wave] = s; sh[8 + wave] = q; }
  __syncthreads();
  if (t == 0) {
    float a = 0.f, c = 0.f;
    for (int i = 0; i < 8; ++i) { a += sh[i]; c += sh[8 + i]; }
    sh[0] = a; sh[8] = c;
  }
  __syncthreads();
  float mu  = sh[0] * (1.f / C_DIM);
  float var = sh[8] * (1.f / C_DIM) - mu * mu;
  float o = g[t] * (v - mu) * rsqrtf(var + 1e-5f) + b[t];
  out[(size_t)row * C_DIM + t] = o;
  if (out2) out2[(size_t)row * C_DIM + t] = o;
}

// =====================================================================
// Tiny batch-attention: scores(n,h,i,j)=q_i.k_j/sqrt(DH), i,j over B=4
// One wave per (n,h); lane = d (DH=32).
// =====================================================================
__global__ __launch_bounds__(256)
void batch_attn_kernel(const float* __restrict__ qkbuf,
                       const float* __restrict__ vbuf,
                       float* __restrict__ attnout) {
  int gid = blockIdx.x * 8 + (threadIdx.x >> 5);
  if (gid >= NQ * HEADS) return;
  int lane = threadIdx.x & 31;
  int n = gid / HEADS, h = gid % HEADS;
  float qv[BATCH], kv[BATCH], vv[BATCH];
#pragma unroll
  for (int i = 0; i < BATCH; ++i) {
    size_t base = ((size_t)i * NQ + n) * 512 + h * DH + lane;
    qv[i] = qkbuf[base];
    kv[i] = qkbuf[base + 256];
    vv[i] = vbuf[((size_t)i * NQ + n) * C_DIM + h * DH + lane];
  }
  const float scale = 0.17677669529663687f;  // 1/sqrt(32)
  float sc[BATCH][BATCH];
#pragma unroll
  for (int i = 0; i < BATCH; ++i)
#pragma unroll
    for (int j = 0; j < BATCH; ++j)
      sc[i][j] = wred32(qv[i] * kv[j]) * scale;   // broadcast to all lanes
#pragma unroll
  for (int i = 0; i < BATCH; ++i) {
    float m = sc[i][0];
#pragma unroll
    for (int j = 1; j < BATCH; ++j) m = fmaxf(m, sc[i][j]);
    float e[BATCH], s = 0.f;
#pragma unroll
    for (int j = 0; j < BATCH; ++j) { e[j] = __expf(sc[i][j] - m); s += e[j]; }
    float inv = 1.f / s, o = 0.f;
#pragma unroll
    for (int j = 0; j < BATCH; ++j) o += e[j] * inv * vv[j];
    attnout[((size_t)i * NQ + n) * C_DIM + h * DH + lane] = o;
  }
}

// =====================================================================
// MS-deformable bilinear sampling. One wave per (b,n,h); lane = d.
// =====================================================================
__global__ __launch_bounds__(256)
void msdeform_sample_kernel(const unsigned short* __restrict__ vbf,
                            const float* __restrict__ offb,
                            const float* __restrict__ awb,
                            const float* __restrict__ ref,
                            float* __restrict__ out) {
  int gid = blockIdx.x * 8 + (threadIdx.x >> 5);
  if (gid >= BATCH * NQ * HEADS) return;
  int lane = threadIdx.x & 31;
  int b = gid / (NQ * HEADS);
  int rem = gid % (NQ * HEADS);
  int n = rem / HEADS, h = rem % HEADS;
  size_t row = (size_t)b * NQ + n;
  float rx = ref[row * 3 + 0], ry = ref[row * 3 + 1];

  float w[NPTS], m = -1e30f;
#pragma unroll
  for (int p = 0; p < NPTS; ++p) {
    w[p] = awb[row * 64 + h * NPTS + p];
    m = fmaxf(m, w[p]);
  }
  float s = 0.f;
#pragma unroll
  for (int p = 0; p < NPTS; ++p) { w[p] = __expf(w[p] - m); s += w[p]; }
  float inv = 1.f / s;

  float acc = 0.f;
  const unsigned short* vb = vbf + (size_t)b * HW * C_DIM + h * DH + lane;
#pragma unroll
  for (int p = 0; p < NPTS; ++p) {
    float ox = offb[row * 128 + (h * NPTS + p) * 2 + 0];
    float oy = offb[row * 128 + (h * NPTS + p) * 2 + 1];
    float x = (rx + ox * (1.f / WFM)) * WFM - 0.5f;
    float y = (ry + oy * (1.f / HFM)) * HFM - 0.5f;
    float x0f = floorf(x), y0f = floorf(y);
    float dx = x - x0f, dy = y - y0f;
    int x0 = (int)x0f, y0 = (int)y0f;
    float pw = w[p] * inv, g = 0.f;
#pragma unroll
    for (int cy = 0; cy < 2; ++cy)
#pragma unroll
      for (int cx = 0; cx < 2; ++cx) {
        int xi = x0 + cx, yi = y0 + cy;
        float bw = (cy ? dy : 1.f - dy) * (cx ? dx : 1.f - dx);
        float valid =
            (xi >= 0 && xi < WFM && yi >= 0 && yi < HFM) ? 1.f : 0.f;
        int xc = min(max(xi, 0), WFM - 1);
        int yc = min(max(yi, 0), HFM - 1);
        g += bf2f(vb[(size_t)(yc * WFM + xc) * C_DIM]) * bw * valid;
      }
    acc += g * pw;
  }
  out[row * C_DIM + h * DH + lane] = acc;
}

// =====================================================================
// copy reference_points into the output tuple (refs + 6 stacked copies)
// =====================================================================
__global__ void copy_refs_kernel(const float* __restrict__ ref,
                                 float* __restrict__ out) {
  int i = blockIdx.x * blockDim.x + threadIdx.x;
  const int NREF = BATCH * NQ * 3;          // 10800
  if (i >= NREF) return;
  float v = ref[i];
  const size_t base = (size_t)NLAYER * ROWS * C_DIM;   // 5,529,600
  out[base + i] = v;
  for (int l = 0; l < NLAYER; ++l)
    out[base + NREF + (size_t)l * NREF + i] = v;
}

// =====================================================================
// host side
// =====================================================================
static void launch_gemm(const float* A, const unsigned short* W,
                        const float* bias, const float* res, void* out,
                        int M, int N, int K, bool outbf, bool relu,
                        hipStream_t s) {
  dim3 g(N / 64, (M + 255) / 256), b(256);
  if (outbf)
    gemm_bf16_kernel<1, 0, 0><<<g, b, 0, s>>>(A, W, bias, nullptr, out, M, N, K);
  else if (relu)
    gemm_bf16_kernel<0, 1, 0><<<g, b, 0, s>>>(A, W, bias, nullptr, out, M, N, K);
  else if (res)
    gemm_bf16_kernel<0, 0, 1><<<g, b, 0, s>>>(A, W, bias, res, out, M, N, K);
  else
    gemm_bf16_kernel<0, 0, 0><<<g, b, 0, s>>>(A, W, bias, nullptr, out, M, N, K);
}

extern "C" void kernel_launch(void* const* d_in, const int* in_sizes, int n_in,
                              void* d_out, int out_size, void* d_ws,
                              size_t ws_size, hipStream_t stream) {
  const float* query     = (const float*)d_in[0];
  const float* query_pos = (const float*)d_in[1];
  const float* value     = (const float*)d_in[2];
  const float* refpts    = (const float*)d_in[3];
  const float* sa_wqkv  = (const float*)d_in[4];
  const float* sa_bqkv  = (const float*)d_in[5];
  const float* sa_wo    = (const float*)d_in[6];
  const float* sa_bo    = (const float*)d_in[7];
  const float* off_w    = (const float*)d_in[8];
  const float* off_b    = (const float*)d_in[9];
  const float* aw_w     = (const float*)d_in[10];
  const float* aw_b     = (const float*)d_in[11];
  const float* vproj_w  = (const float*)d_in[12];
  const float* vproj_b  = (const float*)d_in[13];
  const float* oproj_w  = (const float*)d_in[14];
  const float* oproj_b  = (const float*)d_in[15];
  const float* mlp_w    = (const float*)d_in[16];
  const float* mlp_b    = (const float*)d_in[17];
  const float* ffn_w1   = (const float*)d_in[18];
  const float* ffn_b1   = (const float*)d_in[19];
  const float* ffn_w2   = (const float*)d_in[20];
  const float* ffn_b2   = (const float*)d_in[21];
  const float* n1_g = (const float*)d_in[22];
  const float* n1_b = (const float*)d_in[23];
  const float* n2_g = (const float*)d_in[24];
  const float* n2_b = (const float*)d_in[25];
  const float* n3_g = (const float*)d_in[26];
  const float* n3_b = (const float*)d_in[27];

  // ---- workspace layout ----
  char* base = (char*)d_ws;
  size_t off = 0;
  auto alloc = [&](size_t bytes) {
    size_t o = off; off += (bytes + 255) & ~(size_t)255; return o;
  };
  unsigned short* wqkv_bf = (unsigned short*)(base + alloc((size_t)6*768*256*2));
  unsigned short* wo_bf   = (unsigned short*)(base + alloc((size_t)6*256*256*2));
  unsigned short* offw_bf = (unsigned short*)(base + alloc((size_t)6*128*256*2));
  unsigned short* aww_bf  = (unsigned short*)(base + alloc((size_t)6*64*256*2));
  unsigned short* vpr_bf  = (unsigned short*)(base + alloc((size_t)6*256*256*2));
  unsigned short* opr_bf  = (unsigned short*)(base + alloc((size_t)6*256*256*2));
  unsigned short* mlp_bf  = (unsigned short*)(base + alloc((size_t)6*256*256*2));
  unsigned short* f1_bf   = (unsigned short*)(base + alloc((size_t)6*512*256*2));
  unsigned short* f2_bf   = (unsigned short*)(base + alloc((size_t)6*256*512*2));
  float* value_flat = (float*)(base + alloc((size_t)VROWS*C_DIM*4));
  unsigned short* vbf = (unsigned short*)(base + alloc((size_t)VROWS*C_DIM*2));
  float* qcur   = (float*)(base + alloc((size_t)ROWS*C_DIM*4));
  float* qpp0   = (float*)(base + alloc((size_t)ROWS*C_DIM*4));
  float* qkbuf  = (float*)(base + alloc((size_t)ROWS*512*4));
  float* vbuf   = (float*)(base + alloc((size_t)ROWS*C_DIM*4));
  float* attnout= (float*)(base + alloc((size_t)ROWS*C_DIM*4));
  float* q1     = (float*)(base + alloc((size_t)ROWS*C_DIM*4));
  float* qpp    = (float*)(base + alloc((size_t)ROWS*C_DIM*4));
  float* offb   = (float*)(base + alloc((size_t)ROWS*128*4));
  float* awb    = (float*)(base + alloc((size_t)ROWS*64*4));
  float* dsamp  = (float*)(base + alloc((size_t)ROWS*C_DIM*4));
  float* doproj = (float*)(base + alloc((size_t)ROWS*C_DIM*4));
  float* mlpo   = (float*)(base + alloc((size_t)ROWS*C_DIM*4));
  float* q2     = (float*)(base + alloc((size_t)ROWS*C_DIM*4));
  float* hbuf   = (float*)(base + alloc((size_t)ROWS*512*4));
  float* ffn2o  = (float*)(base + alloc((size_t)ROWS*C_DIM*4));
  (void)ws_size; (void)in_sizes; (void)n_in; (void)out_size;

  float* out = (float*)d_out;

  // ---- one-time prep ----
  auto cvt = [&](const float* src, unsigned short* dst, size_t n) {
    cvt_bf16_kernel<<<(unsigned)((n + 255) / 256), 256, 0, stream>>>(src, dst, n);
  };
  cvt(sa_wqkv, wqkv_bf, (size_t)6*768*256);
  cvt(sa_wo,   wo_bf,   (size_t)6*256*256);
  cvt(off_w,   offw_bf, (size_t)6*128*256);
  cvt(aw_w,    aww_bf,  (size_t)6*64*256);
  cvt(vproj_w, vpr_bf,  (size_t)6*256*256);
  cvt(oproj_w, opr_bf,  (size_t)6*256*256);
  cvt(mlp_w,   mlp_bf,  (size_t)6*256*256);
  cvt(ffn_w1,  f1_bf,   (size_t)6*512*256);
  cvt(ffn_w2,  f2_bf,   (size_t)6*256*512);

  transpose_value_kernel<<<dim3(HW / 32, C_DIM / 32, BATCH), dim3(32, 8), 0,
                           stream>>>(value, value_flat);
  copy_refs_kernel<<<(BATCH * NQ * 3 + 255) / 256, 256, 0, stream>>>(refpts, out);
  hipMemcpyAsync(qcur, query, (size_t)ROWS * C_DIM * 4,
                 hipMemcpyDeviceToDevice, stream);

  const int NE = ROWS * C_DIM;
  for (int l = 0; l < NLAYER; ++l) {
    // --- self attention ---
    add_kernel<<<(NE + 255) / 256, 256, 0, stream>>>(qcur, query_pos, qpp0, NE);
    launch_gemm(qpp0, wqkv_bf + (size_t)l * 768 * 256, sa_bqkv + l * 768,
                nullptr, qkbuf, ROWS, 512, 256, false, false, stream);
    launch_gemm(qcur, wqkv_bf + (size_t)l * 768 * 256 + (size_t)512 * 256,
                sa_bqkv + l * 768 + 512, nullptr, vbuf, ROWS, 256, 256, false,
                false, stream);
    batch_attn_kernel<<<(NQ * HEADS) / 8, 256, 0, stream>>>(qkbuf, vbuf, attnout);
    launch_gemm(attnout, wo_bf + (size_t)l * 256 * 256, sa_bo + l * 256,
                nullptr, qpp0 /*tgt scratch*/, ROWS, 256, 256, false, false,
                stream);
    ln_kernel<<<ROWS, 256, 0, stream>>>(qcur, qpp0, n1_g + l * 256,
                                        n1_b + l * 256, q1, nullptr);

    // --- ms-deform attention ---
    add_kernel<<<(NE + 255) / 256, 256, 0, stream>>>(q1, query_pos, qpp, NE);
    launch_gemm(qpp, offw_bf + (size_t)l * 128 * 256, off_b + l * 128, nullptr,
                offb, ROWS, 128, 256, false, false, stream);
    launch_gemm(qpp, aww_bf + (size_t)l * 64 * 256, aw_b + l * 64, nullptr,
                awb, ROWS, 64, 256, false, false, stream);
    launch_gemm(value_flat, vpr_bf + (size_t)l * 256 * 256, vproj_b + l * 256,
                nullptr, vbf, VROWS, 256, 256, true /*bf16 out*/, false, stream);
    msdeform_sample_kernel<<<(BATCH * NQ * HEADS) / 8, 256, 0, stream>>>(
        vbf, offb, awb, refpts, dsamp);
    launch_gemm(dsamp, opr_bf + (size_t)l * 256 * 256, oproj_b + l * 256,
                nullptr, doproj, ROWS, 256, 256, false, false, stream);
    launch_gemm(doproj, mlp_bf + (size_t)l * 256 * 256, mlp_b + l * 256,
                q1 /*residual*/, mlpo, ROWS, 256, 256, false, false, stream);
    ln_kernel<<<ROWS, 256, 0, stream>>>(mlpo, nullptr, n2_g + l * 256,
                                        n2_b + l * 256, q2, nullptr);

    // --- FFN ---
    launch_gemm(q2, f1_bf + (size_t)l * 512 * 256, ffn_b1 + l * 512, nullptr,
                hbuf, ROWS, 512, 256, false, true /*relu*/, stream);
    launch_gemm(hbuf, f2_bf + (size_t)l * 256 * 512, ffn_b2 + l * 256,
                q2 /*residual*/, ffn2o, ROWS, 256, 512, false, false, stream);
    ln_kernel<<<ROWS, 256, 0, stream>>>(ffn2o, nullptr, n3_g + l * 256,
                                        n3_b + l * 256, qcur,
                                        out + (size_t)l * ROWS * C_DIM);
  }
}
